// LRSVConv_62989990363676
// MI455X (gfx1250) — matrix operand order repaired and checked by reading, not compiled, not run on hardware
//
#include <hip/hip_runtime.h>
#include <hip/hip_bf16.h>

// LRSV fused conv3x3 + rank-weighted reduction for MI455X (gfx1250, wave32).
// Pass 1: convert conv_w fp32 -> f16 in [oc][tap][cin] layout in d_ws.
// Pass 2: implicit GEMM, M=192 out-channels, N=pixels, K=288 (9 taps x 32 cin),
//         f16 WMMA 16x16x32 (fp32 accum); weights DMA'd to LDS via
//         global_load_async_to_lds_b128 (ASYNCcnt); x tile double-buffered in
//         LDS (one barrier per chunk); rank-combine fused in the epilogue.

typedef __attribute__((ext_vector_type(16))) _Float16 v16h;
typedef __attribute__((ext_vector_type(8)))  _Float16 v8h;
typedef __attribute__((ext_vector_type(2)))  _Float16 v2h;
typedef __attribute__((ext_vector_type(8)))  float    v8f;

#define NB    8
#define CIN   32
#define COUT  64
#define OCH   192      // COUT * RANK
#define IMG   256
#define IMG2  (IMG * IMG)
#define KTOT  288      // CIN * 9
#define WS_HALVES (OCH * KTOT)          // 55296 f16  = 110592 B
#define WS_BYTES  (WS_HALVES * 2)
#define XS_HALVES (3 * 66 * CIN)        // 6336  f16  = 12672 B
#define XS_BYTES  (XS_HALVES * 2)
#define SMEM_BYTES (WS_BYTES + 2 * XS_BYTES)   // 135936 B
#define WS_B128   (WS_BYTES / 16)       // 6912 = 27 * 256

// ---- Pass 1: fp32 -> f16 weight transpose into d_ws -----------------------
__global__ __launch_bounds__(256) void
prep_weights(const float* __restrict__ w, _Float16* __restrict__ wh) {
    int d = blockIdx.x * 256 + threadIdx.x;      // dest index, coalesced b16
    if (d < WS_HALVES) {
        int oc  = d / KTOT;
        int rem = d - oc * KTOT;
        int tap = rem >> 5;
        int cin = rem & 31;
        wh[d] = (_Float16)w[(oc * CIN + cin) * 9 + tap];
    }
}

// ---- Pass 2: fused conv + rank reduction ----------------------------------
__global__ __launch_bounds__(256) void
lrsv_conv_wmma(const float* __restrict__ x,
               const _Float16* __restrict__ wh,
               const float* __restrict__ kw,
               float* __restrict__ out) {
    extern __shared__ __align__(16) char smem[];
    _Float16* ws   = (_Float16*)smem;               // [192][9][32] f16, LDS off 0
    _Float16* xsb0 = (_Float16*)(smem + WS_BYTES);
    _Float16* xsb1 = (_Float16*)(smem + WS_BYTES + XS_BYTES);

    const int blk  = blockIdx.x;        // 2048 blocks: one image row each
    const int b    = blk >> 8;
    const int y    = blk & 255;

    const int tid   = threadIdx.x;
    const int lane  = tid & 31;
    const int wave  = tid >> 5;         // 0..7
    const int wj    = wave & 3;         // channel group (16 oc)
    const int wp    = wave >> 2;        // pixel half (32 px)
    const int lhalf = lane >> 4;        // 0/1
    const int l16   = lane & 15;

    // ---- stage all f16 weights into LDS via async DMA (no VGPR roundtrip).
    // No static __shared__, so dynamic LDS begins at offset 0: byte offset
    // i*16 is both the wh offset and the LDS VDST address. Uniform trip count
    // (6912 = 27*256) -> no exec-masked loop.
    #pragma unroll
    for (int it = 0; it < WS_B128 / 256; ++it) {
        int off = (it * 256 + tid) * 16;
        asm volatile("global_load_async_to_lds_b128 %0, %1, %2 offset:0"
                     :: "v"(off), "v"(off), "s"(wh)
                     : "memory");
    }
    asm volatile("s_wait_asynccnt 0x0" ::: "memory");

    // ---- per-thread x-staging coordinates (fixed across chunks) ----
    // 198 active threads; each owns one (row,col) halo position, all 32 cin.
    const int  srow = tid / 66;                 // 0..3 (>=3 masked off)
    const int  scol = tid - srow * 66;          // 0..65
    const int  gy   = y + srow - 1;
    const bool sact = (tid < 198) && ((unsigned)gy < 256u);
    const float* srcb = x + ((size_t)(b * CIN) * IMG + gy) * IMG; // + c*IMG2 + gx

    // self-contained staging phase: 32 loads in flight (imm offsets off one
    // base), pack-convert, 4x ds_store_b128. Nothing stays live past here.
    auto stage = [&](int x0c, _Float16* buf) {
        int  gx = x0c + scol - 1;
        bool ok = sact && ((unsigned)gx < 256u);
        const float* src = srcb + gx;
        float f[32];
        #pragma unroll
        for (int c = 0; c < 32; ++c)
            f[c] = ok ? src[c * IMG2] : 0.f;    // imm offset c*256KB < 2^23
        if (tid < 198) {
            v2h p[16];
            #pragma unroll
            for (int c = 0; c < 16; ++c) {
                auto pk = __builtin_amdgcn_cvt_pkrtz(f[2 * c], f[2 * c + 1]);
                p[c] = __builtin_bit_cast(v2h, pk);
            }
            v8h* dst = (v8h*)&buf[(srow * 66 + scol) * CIN];
            #pragma unroll
            for (int q = 0; q < 4; ++q)         // 4x ds_store_b128, contiguous
                dst[q] = ((const v8h*)p)[q];
        }
    };

    stage(0, xsb0);

    const v8f vzero = {};
    const float* kwy = kw + y * IMG;            // + px, + IMG2 + px (imm)
    // per-thread output base: element (nn,vr) at ob[vr*IMG2 + nn*16] (imm offs)
    float* obase = out + (((size_t)(b * COUT + wj * 16 + 8 * lhalf) * IMG + y) * IMG);

    for (int chunk = 0; chunk < 4; ++chunk) {
        const int x0 = chunk * 64;
        __syncthreads();   // prev chunk fully consumed; this chunk's buf ready

        if (chunk < 3) stage(x0 + 64, (chunk & 1) ? xsb0 : xsb1);

        const _Float16* xsr = (chunk & 1) ? xsb1 : xsb0;

        v8f acc[2][3];
        #pragma unroll
        for (int nn = 0; nn < 2; ++nn)
            #pragma unroll
            for (int r = 0; r < 3; ++r)
                acc[nn][r] = vzero;

        #pragma unroll
        for (int tap = 0; tap < 9; ++tap) {
            const int dy = tap / 3;             // 0..2 == LDS row
            const int dx = tap % 3;

            // B fragments (x): lane n = pixel l16, halves 0-7: K = 8*lhalf+i ; halves 8-15: K = 16+8*lhalf+i
            v16h bfr[2];
            #pragma unroll
            for (int nn = 0; nn < 2; ++nn) {
                const _Float16* bp =
                    &xsr[(dy * 66 + (wp * 32 + nn * 16 + l16 + dx)) * CIN + 8 * lhalf];
                v8h blo = *(const v8h*)(bp);
                v8h bhi = *(const v8h*)(bp + 16);
                bfr[nn] = __builtin_shufflevector(
                    blo, bhi, 0,1,2,3,4,5,6,7,8,9,10,11,12,13,14,15);
            }

            #pragma unroll
            for (int r = 0; r < 3; ++r) {       // A shared by both B fragments
                // A fragment (weights): lane m = oc row l16
                // halves 0-7: K = 8*lhalf+i ; halves 8-15: K = 16+8*lhalf+i
                int oc = (wj + 4 * r) * 16 + l16;
                const _Float16* ap = &ws[(oc * 9 + tap) * CIN + 8 * lhalf];
                v8h alo = *(const v8h*)(ap);
                v8h ahi = *(const v8h*)(ap + 16);
                v16h afr = __builtin_shufflevector(
                    alo, ahi, 0,1,2,3,4,5,6,7,8,9,10,11,12,13,14,15);
                acc[0][r] = __builtin_amdgcn_wmma_f32_16x16x32_f16(
                    false, afr, false, bfr[0], (short)0, acc[0][r], false, false);
                acc[1][r] = __builtin_amdgcn_wmma_f32_16x16x32_f16(
                    false, afr, false, bfr[1], (short)0, acc[1][r], false, false);
            }
        }

        // ---- epilogue: out = acc_r0 + kw0*acc_r1 + kw1*acc_r2 ----
        const int pxw = x0 + wp * 32 + l16;     // pixel of nn=0
        float* ob = obase + pxw;
        #pragma unroll
        for (int nn = 0; nn < 2; ++nn) {
            const float k0 = kwy[pxw + nn * 16];
            const float k1 = kwy[IMG2 + pxw + nn * 16];
            #pragma unroll
            for (int vr = 0; vr < 8; ++vr) {
                float v = acc[nn][0][vr] + k0 * acc[nn][1][vr]
                                         + k1 * acc[nn][2][vr];
                ob[vr * IMG2 + nn * 16] = v;    // imm offsets off one base
            }
        }
    }
}

extern "C" void kernel_launch(void* const* d_in, const int* in_sizes, int n_in,
                              void* d_out, int out_size, void* d_ws, size_t ws_size,
                              hipStream_t stream) {
    (void)in_sizes; (void)n_in; (void)out_size; (void)ws_size;
    const float* x  = (const float*)d_in[0];
    const float* w  = (const float*)d_in[1];
    const float* kw = (const float*)d_in[2];
    float* out      = (float*)d_out;
    _Float16* wh    = (_Float16*)d_ws;           // 110592 B staging

    prep_weights<<<dim3((WS_HALVES + 255) / 256), dim3(256), 0, stream>>>(w, wh);

    dim3 grid(NB * IMG);   // 2048 workgroups: one (batch,row) each
    dim3 block(256);       // 8 wave32
    lrsv_conv_wmma<<<grid, block, SMEM_BYTES, stream>>>(x, wh, kw, out);
}
